// TritonSpatialMLP_19069654794880
// MI455X (gfx1250) — compile-verified, matrix operand order checked by reading, not run on hardware
//
#include <hip/hip_runtime.h>

// ---------------------------------------------------------------------------
// Swin per-head grouped GEMM:  out[t, n, :] = x[t, n, :] @ W[n]   (fp32 in/out)
//   T = 8*56*56 = 25088 tokens, 8 heads, d = 256.
// Strategy: bf16x3 split-precision GEMM on the CDNA5 WMMA pipe
//   (x_hi*W_hi + x_lo*W_hi + x_hi*W_lo), W staged in LDS pre-swizzled into the
//   v_wmma_f32_16x16x32_bf16 B-fragment layout.
// ---------------------------------------------------------------------------

typedef __attribute__((ext_vector_type(16))) __bf16 v16bf;
typedef __attribute__((ext_vector_type(8)))  float  v8f;

#define C_DIM     2048
#define D_DIM     256
#define N_HEADS   8
#define TOKENS    25088
#define BLK_TOK   128
#define LDS_PLANE 65536                       // bf16 elements per plane (256*256)
#define LDS_BYTES (2 * LDS_PLANE * 2)         // hi + lo planes = 256 KiB

// Load 16 fp32 A elements for this lane (two 32B chunks at +0 and +16 elems,
// matching the 16-bit A 16x32 fragment layout) and split into hi/lo bf16.
__device__ __forceinline__ void split_load16(const float* __restrict__ p,
                                             v16bf& hi, v16bf& lo) {
  float4 f0 = *(const float4*)(p + 0);
  float4 f1 = *(const float4*)(p + 4);
  float4 f2 = *(const float4*)(p + 16);
  float4 f3 = *(const float4*)(p + 20);
  float v[16] = {f0.x, f0.y, f0.z, f0.w, f1.x, f1.y, f1.z, f1.w,
                 f2.x, f2.y, f2.z, f2.w, f3.x, f3.y, f3.z, f3.w};
  union { v16bf v; __bf16 e[16]; } uh, ul;
#pragma unroll
  for (int i = 0; i < 16; ++i) {
    __bf16 h = (__bf16)v[i];              // RNE f32 -> bf16
    uh.e[i] = h;
    ul.e[i] = (__bf16)(v[i] - (float)h);  // residual
  }
  hi = uh.v;
  lo = ul.v;
}

extern "C" __global__ void __launch_bounds__(256, 1)
swin_headgemm_bf16x3(const float* __restrict__ x,
                     const float* __restrict__ w,
                     float* __restrict__ out) {
  extern __shared__ __align__(32) char smem[];
  __bf16* lds_hi = (__bf16*)smem;
  __bf16* lds_lo = lds_hi + LDS_PLANE;

  const int head = blockIdx.y;
  const int tok0 = blockIdx.x * BLK_TOK;
  const int tid  = threadIdx.x;

  // ---- Stage W[head] (256x256 fp32) into LDS as hi/lo bf16 planes,
  //      pre-swizzled into the WMMA B (32x16 bf16) fragment layout:
  //      per 32x16 tile: lane = ((k>>3)&1)*16 + n, per-lane 32B chunk holds
  //      VGPR0..7 pairs:  vgpr = ((k>>4)&1)*4 + ((k>>1)&3), half = k&1.
  {
    const float* Wg = w + (size_t)head * (D_DIM * D_DIM) + tid; // column e = tid
    const int nt = tid >> 4, nn = tid & 15;
#pragma unroll 4
    for (int k = 0; k < D_DIM; ++k) {
      float  wv = Wg[(size_t)k * D_DIM];            // coalesced across tid
      __bf16 h  = (__bf16)wv;
      __bf16 l  = (__bf16)(wv - (float)h);
      const int kk  = k & 31;
      const int off = ((k >> 5) * 16 + nt) * 512                       // tile base (ushorts)
                    + ((((kk >> 3) & 1) << 4) + nn) * 16               // lane * 32B
                    + ((((kk >> 4) & 1) << 2) + ((kk >> 1) & 3)) * 2   // vgpr * 4B
                    + (kk & 1);                                        // half
      lds_hi[off] = h;
      lds_lo[off] = l;
    }
  }
  __syncthreads();

  // ---- Compute: 8 waves = 4 M-groups (32 tokens) x 2 N-halves (128 cols).
  const int wave = tid >> 5;
  const int lane = tid & 31;
  const int mgrp = wave >> 1;   // 0..3
  const int ngrp = wave & 1;    // 0..1
  const int m    = lane & 15;   // A-fragment row
  const int g    = lane >> 4;   // half-wave group

  const float* xr0 = x + ((size_t)(tok0 + mgrp * 32 + m) * C_DIM + head * D_DIM);
  const float* xr1 = xr0 + (size_t)16 * C_DIM;

  v8f acc[2][8] = {};

  for (int ks = 0; ks < 8; ++ks) {        // K = 256 in steps of 32
    const int kb = ks * 32 + g * 8;
    v16bf ah0, al0, ah1, al1;
    split_load16(xr0 + kb, ah0, al0);
    split_load16(xr1 + kb, ah1, al1);
    if (ks < 7) {                         // speculative prefetch of next A chunk
      __builtin_prefetch(xr0 + kb + 32, 0, 0);
      __builtin_prefetch(xr1 + kb + 32, 0, 0);
    }
    const __bf16* bh = lds_hi + (ks * 16 + ngrp * 8) * 512 + lane * 16;
    const __bf16* bl = lds_lo + (ks * 16 + ngrp * 8) * 512 + lane * 16;
#pragma unroll
    for (int nt = 0; nt < 8; ++nt) {
      v16bf Bh = *(const v16bf*)(bh + nt * 512);
      v16bf Bl = *(const v16bf*)(bl + nt * 512);
      // D = A_hi*B_hi + A_lo*B_hi + A_hi*B_lo + C  (bf16x3 emulated fp32)
      acc[0][nt] = __builtin_amdgcn_wmma_f32_16x16x32_bf16(false, ah0, false, Bh,
                     (short)0, acc[0][nt], false, false);
      acc[1][nt] = __builtin_amdgcn_wmma_f32_16x16x32_bf16(false, ah1, false, Bh,
                     (short)0, acc[1][nt], false, false);
      acc[0][nt] = __builtin_amdgcn_wmma_f32_16x16x32_bf16(false, al0, false, Bh,
                     (short)0, acc[0][nt], false, false);
      acc[1][nt] = __builtin_amdgcn_wmma_f32_16x16x32_bf16(false, al1, false, Bh,
                     (short)0, acc[1][nt], false, false);
      acc[0][nt] = __builtin_amdgcn_wmma_f32_16x16x32_bf16(false, ah0, false, Bl,
                     (short)0, acc[0][nt], false, false);
      acc[1][nt] = __builtin_amdgcn_wmma_f32_16x16x32_bf16(false, ah1, false, Bl,
                     (short)0, acc[1][nt], false, false);
    }
  }

  // ---- Epilogue: D layout = VGPR r: lanes 0-15 -> M=r, lanes 16-31 -> M=r+8;
  //      N = lane & 15. Streaming output -> non-temporal stores.
  const int nn = lane & 15;
  float* ob = out + ((size_t)(tok0 + mgrp * 32) * C_DIM + head * D_DIM + ngrp * 128);
#pragma unroll
  for (int mt = 0; mt < 2; ++mt) {
#pragma unroll
    for (int nt = 0; nt < 8; ++nt) {
      union { v8f v; float e[8]; } u;
      u.v = acc[mt][nt];
#pragma unroll
      for (int r = 0; r < 8; ++r) {
        __builtin_nontemporal_store(
            u.e[r], ob + (size_t)(mt * 16 + g * 8 + r) * C_DIM + nt * 16 + nn);
      }
    }
  }
}

extern "C" void kernel_launch(void* const* d_in, const int* in_sizes, int n_in,
                              void* d_out, int out_size, void* d_ws, size_t ws_size,
                              hipStream_t stream) {
  (void)in_sizes; (void)n_in; (void)out_size; (void)d_ws; (void)ws_size;
  const float* x = (const float*)d_in[0];
  const float* w = (const float*)d_in[1];
  float* out     = (float*)d_out;

  // 256 KiB dynamic LDS per workgroup (<= 320 KiB per WGP on CDNA5).
  (void)hipFuncSetAttribute(reinterpret_cast<const void*>(&swin_headgemm_bf16x3),
                            hipFuncAttributeMaxDynamicSharedMemorySize, LDS_BYTES);

  dim3 grid(TOKENS / BLK_TOK, N_HEADS, 1);   // 196 x 8 workgroups
  swin_headgemm_bf16x3<<<grid, dim3(256, 1, 1), LDS_BYTES, stream>>>(x, w, out);
}
